// ScaledQuantizedLinearINT4_2061584302889
// MI455X (gfx1250) — matrix operand order repaired and checked by reading, not compiled, not run on hardware
//
#include <hip/hip_runtime.h>
#include <hip/hip_bf16.h>
#include <math.h>

typedef __attribute__((ext_vector_type(16))) _Float16 v16h;
typedef __attribute__((ext_vector_type(8)))  _Float16 v8h;
typedef __attribute__((ext_vector_type(8)))  float    v8f;

#define BATCH 64
#define INF   4096
#define OUTF  8192
#define KC    16
#define HISTB 2048
#define SUMSCALE 65536.0f

__device__ __forceinline__ float signed_pow_f(float v, float a) {
  return copysignf(powf(fabsf(v), a), v);
}
// order-preserving float<->uint encoding for deterministic atomic min/max
__device__ __forceinline__ unsigned fenc(float f) {
  unsigned u = __float_as_uint(f);
  return (u & 0x80000000u) ? ~u : (u | 0x80000000u);
}
__device__ __forceinline__ float fdec(unsigned e) {
  unsigned u = (e & 0x80000000u) ? (e & 0x7FFFFFFFu) : ~e;
  return __uint_as_float(u);
}

// ---- scratch init (ws is poisoned by harness; rebuild every call) ----
__global__ void k_init(unsigned long long* hc, unsigned long long* hs, unsigned* mm) {
  int i = blockIdx.x * blockDim.x + threadIdx.x;
  if (i < HISTB) { hc[i] = 0ull; hs[i] = 0ull; }
  if (i == 0) { mm[0] = 0xFFFFFFFFu; mm[1] = 0u; }
}

// ---- min/max of signed_pow(W, alpha) ----
__global__ void k_minmax(const float* __restrict__ w, const float* __restrict__ alpha,
                         unsigned* mm, int n) {
  float a = alpha[0];
  unsigned lmin = 0xFFFFFFFFu, lmax = 0u;
  for (int i = blockIdx.x * blockDim.x + threadIdx.x; i < n; i += gridDim.x * blockDim.x) {
    unsigned e = fenc(signed_pow_f(w[i], a));
    lmin = min(lmin, e); lmax = max(lmax, e);
  }
  __shared__ unsigned smin[256], smax[256];
  int t = threadIdx.x;
  smin[t] = lmin; smax[t] = lmax;
  __syncthreads();
  for (int s = 128; s > 0; s >>= 1) {
    if (t < s) { smin[t] = min(smin[t], smin[t + s]); smax[t] = max(smax[t], smax[t + s]); }
    __syncthreads();
  }
  if (t == 0) { atomicMin(&mm[0], smin[0]); atomicMax(&mm[1], smax[0]); }
}

// ---- histogram: count + fixed-point value-sum per bin (deterministic int atomics) ----
__global__ void k_hist(const float* __restrict__ w, const float* __restrict__ alpha,
                       const unsigned* __restrict__ mm,
                       unsigned long long* hc, unsigned long long* hs, int n) {
  __shared__ unsigned long long lc[HISTB], ls[HISTB];
  for (int i = threadIdx.x; i < HISTB; i += blockDim.x) { lc[i] = 0ull; ls[i] = 0ull; }
  __syncthreads();
  float a = alpha[0];
  float vmin = fdec(mm[0]), vmax = fdec(mm[1]);
  float scale = (vmax > vmin) ? (float)HISTB / (vmax - vmin) : 0.0f;
  for (int i = blockIdx.x * blockDim.x + threadIdx.x; i < n; i += gridDim.x * blockDim.x) {
    float v = signed_pow_f(w[i], a);
    int b = (int)((v - vmin) * scale);
    b = max(0, min(HISTB - 1, b));
    atomicAdd(&lc[b], 1ull);
    atomicAdd(&ls[b], (unsigned long long)__float2ll_rn(v * SUMSCALE));
  }
  __syncthreads();
  for (int i = threadIdx.x; i < HISTB; i += blockDim.x) {
    if (lc[i]) atomicAdd(&hc[i], lc[i]);
    if (ls[i]) atomicAdd(&hs[i], ls[i]);
  }
}

// ---- 1-D k-means on the histogram: quantile init + 10 Lloyd iterations ----
__global__ void k_kmeans(const unsigned long long* __restrict__ hc,
                         const unsigned long long* __restrict__ hs,
                         const unsigned* __restrict__ mm, float* cm /*[16 cent | 15 mids]*/) {
  if (threadIdx.x != 0 || blockIdx.x != 0) return;
  float vmin = fdec(mm[0]), vmax = fdec(mm[1]);
  float binw = (vmax - vmin) / (float)HISTB;
  float c[KC], s[KC], cnt[KC], mids[KC - 1];
  double ntot = 0.0;
  for (int j = 0; j < HISTB; j++) ntot += (double)hc[j];
  // quantile init: c0[i] ~ order statistic at (i/15)*(N-1)
  {
    double acc = 0.0; int i = 0;
    for (int j = 0; j < HISTB && i < KC; j++) {
      acc += (double)hc[j];
      float bc = vmin + ((float)j + 0.5f) * binw;
      while (i < KC && acc > ((double)i / (double)(KC - 1)) * (ntot - 1.0)) { c[i] = bc; i++; }
    }
    while (i < KC) { c[i] = vmax; i++; }
  }
  for (int it = 0; it < 10; it++) {
    for (int i = 1; i < KC; i++) { // insertion sort
      float key = c[i]; int j = i - 1;
      while (j >= 0 && c[j] > key) { c[j + 1] = c[j]; j--; }
      c[j + 1] = key;
    }
    for (int m = 0; m < KC - 1; m++) mids[m] = 0.5f * (c[m] + c[m + 1]);
    for (int i = 0; i < KC; i++) { s[i] = 0.f; cnt[i] = 0.f; }
    for (int j = 0; j < HISTB; j++) {
      unsigned long long cj = hc[j];
      if (!cj) continue;
      float bc = vmin + ((float)j + 0.5f) * binw;
      int cl = 0;
      for (int m = 0; m < KC - 1; m++) cl += (bc > mids[m]) ? 1 : 0;
      s[cl]   += (float)((long long)hs[j]) * (1.0f / SUMSCALE);
      cnt[cl] += (float)cj;
    }
    for (int i = 0; i < KC; i++) if (cnt[i] > 0.f) c[i] = s[i] / cnt[i]; // else keep sorted c[i]
  }
  for (int i = 1; i < KC; i++) { // final sort
    float key = c[i]; int j = i - 1;
    while (j >= 0 && c[j] > key) { c[j + 1] = c[j]; j--; }
    c[j + 1] = key;
  }
  for (int i = 0; i < KC; i++) cm[i] = c[i];
  for (int m = 0; m < KC - 1; m++) cm[KC + m] = 0.5f * (c[m] + c[m + 1]);
}

// ---- dequantized fp16 weight: Wq[o,k] = centroid[searchsorted(mids, signed_pow(W))] ----
__global__ void k_quant(const float* __restrict__ w, const float* __restrict__ alpha,
                        const float* __restrict__ cm, _Float16* __restrict__ wq, int n) {
  __shared__ float lc[KC], lm[KC - 1];
  if (threadIdx.x < KC) lc[threadIdx.x] = cm[threadIdx.x];
  if (threadIdx.x < KC - 1) lm[threadIdx.x] = cm[KC + threadIdx.x];
  __syncthreads();
  float a = alpha[0];
  for (int i = blockIdx.x * blockDim.x + threadIdx.x; i < n; i += gridDim.x * blockDim.x) {
    float v = signed_pow_f(w[i], a);
    int cl = 0;
#pragma unroll
    for (int m = 0; m < KC - 1; m++) cl += (v > lm[m]) ? 1 : 0;
    wq[i] = (_Float16)lc[cl];
  }
}

// ---- x_scaled in fp16 ----
__global__ void k_scalex(const float* __restrict__ x, const float* __restrict__ alpha,
                         _Float16* __restrict__ xs, int n) {
  float a = alpha[0];
  for (int i = blockIdx.x * blockDim.x + threadIdx.x; i < n; i += gridDim.x * blockDim.x)
    xs[i] = (_Float16)signed_pow_f(x[i], a);
}

// ---- WMMA GEMM: one wave per 16-wide N tile, all 64 M rows (weight streamed once) ----
__global__ void __launch_bounds__(256) k_gemm(const _Float16* __restrict__ xs,
                                              const _Float16* __restrict__ wq,
                                              const float* __restrict__ alpha,
                                              float* __restrict__ out) {
  const int lane  = threadIdx.x & 31;
  const int ntile = blockIdx.x * (blockDim.x >> 5) + (threadIdx.x >> 5);
  if (ntile * 16 >= OUTF) return;            // wave-uniform: EXEC stays all-ones for WMMA
  const int hl = lane & 15;                  // N column (B/C) or M row (A) within tile
  const int hi = lane >> 4;                  // lane-half selector per ISA §7.12.2
  const _Float16* __restrict__ brow = wq + (size_t)(ntile * 16 + hl) * INF; // Wq row n: contiguous K
  const _Float16* __restrict__ arow = xs + (size_t)hl * INF;
  v8f acc[4] = {v8f{}, v8f{}, v8f{}, v8f{}};
  for (int k0 = 0; k0 < INF; k0 += 32) {
    // B 32x16 f16 frag: halves 0-7 -> K = k0 + 8*hi + h ; halves 8-15 -> K = k0 + 16 + 8*hi + h
    const v8h* bp = (const v8h*)(brow + k0 + hi * 8);
    v8h b0 = bp[0], b1 = bp[2];
    v16h bf = __builtin_shufflevector(b0, b1, 0,1,2,3,4,5,6,7,8,9,10,11,12,13,14,15);
    __builtin_prefetch(brow + k0 + 2048, 0, 1);  // stream weight ahead (global_prefetch_b8)
#pragma unroll
    for (int m = 0; m < 4; m++) {
      // A 16x32 f16 frag: halves 0-7 -> K = k0 + 8*hi + h ; halves 8-15 -> K = k0 + 16 + 8*hi + h
      const _Float16* ar = arow + (size_t)(m * 16) * INF + k0;
      v8h a0 = *(const v8h*)(ar + hi * 8);
      v8h a1 = *(const v8h*)(ar + 16 + hi * 8);
      v16h af = __builtin_shufflevector(a0, a1, 0,1,2,3,4,5,6,7,8,9,10,11,12,13,14,15);
      acc[m] = __builtin_amdgcn_wmma_f32_16x16x32_f16(
          /*neg_a=*/false, af, /*neg_b=*/false, bf,
          /*c_mod=*/(short)0, acc[m], /*reuse_a=*/false, /*reuse_b=*/false);
    }
  }
  const float ia = 1.0f / alpha[0];
#pragma unroll
  for (int m = 0; m < 4; m++) {
#pragma unroll
    for (int r = 0; r < 8; r++) {            // C/D: VGPR r -> M = r + 8*hi, N = lane%16
      int row = m * 16 + hi * 8 + r;
      int col = ntile * 16 + hl;
      float v = acc[m][r];
      out[(size_t)row * OUTF + col] = signed_pow_f(v, ia);
    }
  }
}

extern "C" void kernel_launch(void* const* d_in, const int* in_sizes, int n_in,
                              void* d_out, int out_size, void* d_ws, size_t ws_size,
                              hipStream_t stream) {
  (void)in_sizes; (void)n_in; (void)out_size; (void)ws_size;
  const float* x     = (const float*)d_in[0];
  const float* w     = (const float*)d_in[1];
  const float* alpha = (const float*)d_in[2];
  float* out = (float*)d_out;
  char* ws = (char*)d_ws;

  const int NW = OUTF * INF;   // 33,554,432
  const int NX = BATCH * INF;  // 262,144
  size_t off = 0;
  _Float16* wq = (_Float16*)(ws + off);            off += (size_t)NW * 2;   // 64 MB fp16 weight
  _Float16* xs = (_Float16*)(ws + off);            off += (size_t)NX * 2;   // 512 KB fp16 x
  unsigned long long* hc = (unsigned long long*)(ws + off); off += HISTB * 8;
  unsigned long long* hs = (unsigned long long*)(ws + off); off += HISTB * 8;
  unsigned* mm = (unsigned*)(ws + off);            off += 256;
  float*    cm = (float*)(ws + off);               off += 256;

  k_init  <<<(HISTB + 255) / 256, 256, 0, stream>>>(hc, hs, mm);
  k_minmax<<<2048, 256, 0, stream>>>(w, alpha, mm, NW);
  k_hist  <<<1024, 256, 0, stream>>>(w, alpha, mm, hc, hs, NW);
  k_kmeans<<<1, 32, 0, stream>>>(hc, hs, mm, cm);
  k_quant <<<4096, 256, 0, stream>>>(w, alpha, cm, wq, NW);
  k_scalex<<<512, 256, 0, stream>>>(x, alpha, xs, NX);
  k_gemm  <<<(OUTF / 16) / 8, 256, 0, stream>>>(xs, wq, alpha, out);
}
